// Block_23587960389955
// MI455X (gfx1250) — matrix operand order repaired and checked by reading, not compiled, not run on hardware
//
#include <hip/hip_runtime.h>
#include <math.h>

#define EMBED   1024
#define HEADS   16
#define HD      64
#define SEQ     2048
#define BATCH   4
#define TOKENS  (BATCH * SEQ)     /* 8192 */
#define HIDDEN  (4 * EMBED)       /* 4096 */
#define QKV3    (3 * EMBED)       /* 3072 */
#define SPITCH  2052              /* 2048 + 4 floats: de-conflict LDS banks */
#define BPITCH  136               /* 128 + 8 halves: TDM pad -> LDS row pitch */

typedef __attribute__((ext_vector_type(16))) _Float16 v16h;
typedef __attribute__((ext_vector_type(8)))  float    v8f;
typedef __attribute__((ext_vector_type(4)))  unsigned int u32x4;
typedef __attribute__((ext_vector_type(4)))  int      i32x4;
typedef __attribute__((ext_vector_type(8)))  int      i32x8;

union AF { v16h v; uint4 u[2]; _Float16 e[16]; };

// A-fragment: halves 0..7 at p, halves 8..15 at p+16 (ISA 16-bit A 16x32 layout)
static __device__ inline v16h load_a_frag(const _Float16* p) {
  AF f;
  f.u[0] = *(const uint4*)(p);
  f.u[1] = *(const uint4*)(p + 16);
  return f.v;
}
// B-fragment: halves 0..7 at p, halves 8..15 at p+16 (same k order as A-fragment)
static __device__ inline v16h load_b_frag(const _Float16* p) {
  AF f;
  f.u[0] = *(const uint4*)(p);
  f.u[1] = *(const uint4*)(p + 16);
  return f.v;
}

// ---------------------------------------------------------------------------
// TDM: DMA one B panel (64 W-rows x 128 halves, row stride K halves) into LDS.
// D# per CDNA5 ISA.  pad_enable inserts 4 dwords after
// every 64 dwords (= one 128-half row) -> LDS row pitch 136 halves (BPITCH).
// Tracked by TENSORcnt; wave-scoped, issued by wave 0 only.
// ---------------------------------------------------------------------------
static __device__ inline void tdm_load_b_tile(const _Float16* g, _Float16* lds, int K) {
  const unsigned long long ga = (unsigned long long)(size_t)g;
  const unsigned int lo = (unsigned int)(size_t)lds;  // low 32 bits of generic ptr = LDS byte offset
  u32x4 g0;
  g0[0] = 1u;                                              // count=1, user descriptor
  g0[1] = lo;                                              // lds_addr
  g0[2] = (unsigned int)(ga & 0xffffffffull);              // global_addr[31:0]
  g0[3] = (unsigned int)((ga >> 32) & 0x01ffffffull)       // global_addr[56:32]
          | (2u << 30);                                    // type = 2 (image)
  i32x8 g1;
  g1[0] = (int)((1u << 16)     // data_size = 2 bytes
              | (1u << 20)     // pad_enable
              | (5u << 22)     // pad_interval code 5 = 64 dwords
              | (3u << 25));   // pad_amount  code 3 = 4 dwords
  g1[1] = (int)(128u << 16);   // tensor_dim0 = 128       (bits 79:48)
  g1[2] = (int)(64u  << 16);   // tensor_dim1 = 64        (bits 111:80)
  g1[3] = (int)(128u << 16);   // tile_dim0   = 128       (bits 127:112)
  g1[4] = 64;                  // tile_dim1 = 64, tile_dim2 = 0
  g1[5] = K;                   // tensor_dim0_stride lo32 (elements)
  g1[6] = 0;
  g1[7] = 0;
  const i32x4 z4 = (i32x4){0, 0, 0, 0};
#if defined(__clang_major__) && (__clang_major__ >= 23)
  const i32x8 z8 = (i32x8){0, 0, 0, 0, 0, 0, 0, 0};
  __builtin_amdgcn_tensor_load_to_lds(g0, g1, z4, z4, z8, 0);
#else
  __builtin_amdgcn_tensor_load_to_lds(g0, g1, z4, z4, 0);
#endif
}

// ---------------------------------------------------------------------------
// C = A @ W^T.  A: [M,K] f16 row-major.  W: [Nn,K] f16 row-major.
// Block: 128 threads = 4 waves; wave -> 32(M) x 64(N); block -> 128 x 64.
// B panel (64x128 halves) TDM-staged into double-buffered LDS; A direct.
// EPI 0: f16 out.  EPI 1: f32 = acc + bias + resid.  EPI 2: f16 = gelu_erf.
// ---------------------------------------------------------------------------
template <int EPI>
__global__ __launch_bounds__(128) void gemm_f16_wmma(
    const _Float16* __restrict__ A, const _Float16* __restrict__ W,
    int M, int Nn, int K,
    const float* __restrict__ bias, const float* __restrict__ resid,
    void* __restrict__ outp) {
  (void)M;
  __shared__ _Float16 ldsB[2][64 * BPITCH];  // 2 x 17408 B = 34 KB

  const int lane = threadIdx.x & 31;
  const int wv   = threadIdx.x >> 5;
  const int l16  = lane & 15;
  const int hi   = lane >> 4;
  const int m0 = blockIdx.y * 128 + wv * 32;
  const int n0 = blockIdx.x * 64;

  v8f acc[8];
#pragma unroll
  for (int i = 0; i < 8; ++i) acc[i] = (v8f){0.f,0.f,0.f,0.f,0.f,0.f,0.f,0.f};

  const _Float16* ap0 = A + (size_t)(m0 + l16) * K + hi * 8;
  const _Float16* ap1 = A + (size_t)(m0 + 16 + l16) * K + hi * 8;
  const _Float16* wp  = W + (size_t)n0 * K;

  const int nchunks = K >> 7;  // 128-half K chunks
  if (wv == 0) tdm_load_b_tile(wp, ldsB[0], K);

  for (int c = 0; c < nchunks; ++c) {
    if (wv == 0) {
      if (c + 1 < nchunks) {
        tdm_load_b_tile(wp + (size_t)(c + 1) * 128, ldsB[(c + 1) & 1], K);
        __builtin_amdgcn_s_wait_tensorcnt(1);
      } else {
        __builtin_amdgcn_s_wait_tensorcnt(0);
      }
    }
    __syncthreads();  // buf[c&1] visible to all waves

    const _Float16* bt = ldsB[c & 1];
#pragma unroll
    for (int kk = 0; kk < 4; ++kk) {
      const int k0 = c * 128 + kk * 32;
      const v16h a0 = load_a_frag(ap0 + k0);
      const v16h a1 = load_a_frag(ap1 + k0);
#pragma unroll
      for (int ns = 0; ns < 4; ++ns) {
        const v16h b = load_b_frag(bt + (size_t)(ns * 16 + l16) * BPITCH + kk * 32 + hi * 8);
        acc[ns]     = __builtin_amdgcn_wmma_f32_16x16x32_f16(false, a0, false, b, (short)0, acc[ns],     false, false);
        acc[4 + ns] = __builtin_amdgcn_wmma_f32_16x16x32_f16(false, a1, false, b, (short)0, acc[4 + ns], false, false);
      }
    }
    __syncthreads();  // all reads of buf[c&1] done before it is re-filled
  }

#pragma unroll
  for (int ms = 0; ms < 2; ++ms) {
    const int rbase = m0 + ms * 16 + hi * 8;
#pragma unroll
    for (int ns = 0; ns < 4; ++ns) {
      const int cn = n0 + ns * 16 + l16;
      float bv = 0.f;
      if (EPI != 0) bv = bias[cn];
#pragma unroll
      for (int r = 0; r < 8; ++r) {
        const size_t idx = (size_t)(rbase + r) * Nn + cn;
        const float v = acc[ms * 4 + ns][r];
        if (EPI == 0) {
          ((_Float16*)outp)[idx] = (_Float16)v;
        } else if (EPI == 1) {
          ((float*)outp)[idx] = v + bv + resid[idx];
        } else {
          const float t = v + bv;
          ((_Float16*)outp)[idx] = (_Float16)(0.5f * t * (1.f + erff(t * 0.70710678118654752f)));
        }
      }
    }
  }
}

// ---------------------------------------------------------------------------
// LayerNorm over last dim (1024) -> f16 output.  One block (256 thr) per token.
// ---------------------------------------------------------------------------
__global__ __launch_bounds__(256) void ln_to_f16(
    const float* __restrict__ x, const float* __restrict__ g,
    const float* __restrict__ b, _Float16* __restrict__ y) {
  const int t = blockIdx.x;
  const float* xr = x + (size_t)t * EMBED;
  float s = 0.f, s2 = 0.f;
#pragma unroll
  for (int j = 0; j < 4; ++j) {
    const float v = xr[threadIdx.x + j * 256];
    s += v;
    s2 += v * v;
  }
  __shared__ float red[512];
  red[threadIdx.x] = s;
  red[256 + threadIdx.x] = s2;
  __syncthreads();
  for (int off = 128; off > 0; off >>= 1) {
    if ((int)threadIdx.x < off) {
      red[threadIdx.x] += red[threadIdx.x + off];
      red[256 + threadIdx.x] += red[256 + threadIdx.x + off];
    }
    __syncthreads();
  }
  const float mu  = red[0] * (1.f / EMBED);
  const float var = red[256] * (1.f / EMBED) - mu * mu;
  const float inv = rsqrtf(var + 1e-5f);
  _Float16* yr = y + (size_t)t * EMBED;
#pragma unroll
  for (int j = 0; j < 4; ++j) {
    const int i = threadIdx.x + j * 256;
    yr[i] = (_Float16)((xr[i] - mu) * inv * g[i] + b[i]);
  }
}

__global__ void cvt_f32_f16(const float* __restrict__ in, _Float16* __restrict__ out, int n) {
  const int i = blockIdx.x * 256 + threadIdx.x;
  if (i < n) out[i] = (_Float16)in[i];
}

// qkv [8192,3072] f16 -> q,k [B*H, SEQ, HD] and vT [B*H, HD, SEQ]
__global__ void split_qkv(const _Float16* __restrict__ qkv, _Float16* __restrict__ q,
                          _Float16* __restrict__ k, _Float16* __restrict__ vT) {
  const int i = blockIdx.x * 256 + threadIdx.x;
  if (i >= TOKENS * QKV3) return;
  const int m = i / QKV3;
  const int j = i - m * QKV3;
  const int which = j >> 10;
  const int c = j & 1023;
  const int h = c >> 6;
  const int d = c & 63;
  const int bb = m >> 11;
  const int tt = m & 2047;
  const int bh = bb * HEADS + h;
  const _Float16 val = qkv[i];
  if (which == 0)       q[((size_t)bh * SEQ + tt) * HD + d] = val;
  else if (which == 1)  k[((size_t)bh * SEQ + tt) * HD + d] = val;
  else                  vT[((size_t)bh * HD + d) * SEQ + tt] = val;
}

// ---------------------------------------------------------------------------
// Attention for one (b, h, 16-query tile).  Block = 256 thr = 8 waves.
// Phase 1: S[16][2048] = q @ k^T via WMMA (each wave: 256 key cols).
// Phase 2: row softmax (no 1/sqrt(d) scale, matching reference).
// Phase 3: O = P @ V via WMMA, cross-wave LDS f32 reduction, f16 out [B,N,C].
// ---------------------------------------------------------------------------
__global__ __launch_bounds__(256) void attn_kernel(
    const _Float16* __restrict__ q, const _Float16* __restrict__ k,
    const _Float16* __restrict__ vT, _Float16* __restrict__ o) {
  extern __shared__ float smem[];  // S[16][SPITCH] then O[16][64]
  const int bh = blockIdx.z * HEADS + blockIdx.y;
  const int m0 = blockIdx.x * 16;
  const _Float16* qb = q  + (size_t)bh * SEQ * HD;
  const _Float16* kb = k  + (size_t)bh * SEQ * HD;
  const _Float16* vb = vT + (size_t)bh * HD * SEQ;

  const int lane = threadIdx.x & 31;
  const int wv   = threadIdx.x >> 5;
  const int l16  = lane & 15;
  const int hi   = lane >> 4;

  // ---- phase 1: scores ----
  const v16h a0 = load_a_frag(qb + (size_t)(m0 + l16) * HD + 0  + hi * 8);
  const v16h a1 = load_a_frag(qb + (size_t)(m0 + l16) * HD + 32 + hi * 8);
  for (int tile = 0; tile < 16; ++tile) {
    const int key0 = (wv * 16 + tile) * 16;
    const v16h b0 = load_b_frag(kb + (size_t)(key0 + l16) * HD + 0  + hi * 8);
    const v16h b1 = load_b_frag(kb + (size_t)(key0 + l16) * HD + 32 + hi * 8);
    v8f s = (v8f){0.f,0.f,0.f,0.f,0.f,0.f,0.f,0.f};
    s = __builtin_amdgcn_wmma_f32_16x16x32_f16(false, a0, false, b0, (short)0, s, false, false);
    s = __builtin_amdgcn_wmma_f32_16x16x32_f16(false, a1, false, b1, (short)0, s, false, false);
#pragma unroll
    for (int r = 0; r < 8; ++r) smem[(hi * 8 + r) * SPITCH + key0 + l16] = s[r];
  }
  __syncthreads();

  // ---- phase 2: softmax over 2048 keys per row (16 threads per row) ----
  {
    const int r = threadIdx.x >> 4;
    const int c = threadIdx.x & 15;
    float* Sr = smem + r * SPITCH;
    float mx = -3.4e38f;
    for (int j = c; j < SEQ; j += 16) mx = fmaxf(mx, Sr[j]);
#pragma unroll
    for (int off = 1; off < 16; off <<= 1) mx = fmaxf(mx, __shfl_xor(mx, off, 32));
    float sum = 0.f;
    for (int j = c; j < SEQ; j += 16) {
      const float e = __expf(Sr[j] - mx);
      Sr[j] = e;
      sum += e;
    }
#pragma unroll
    for (int off = 1; off < 16; off <<= 1) sum += __shfl_xor(sum, off, 32);
    const float sc = 1.f / sum;
    for (int j = c; j < SEQ; j += 16) Sr[j] *= sc;
  }
  float* O = smem + 16 * SPITCH;
  for (int i = threadIdx.x; i < 16 * 64; i += 256) O[i] = 0.f;
  __syncthreads();

  // ---- phase 3: O = P @ V (each wave: 256-key K-slice, then LDS reduce) ----
  v8f acc[4];
#pragma unroll
  for (int ns = 0; ns < 4; ++ns) acc[ns] = (v8f){0.f,0.f,0.f,0.f,0.f,0.f,0.f,0.f};
  for (int k0 = wv * 256; k0 < (wv + 1) * 256; k0 += 32) {
    const float* sp = smem + l16 * SPITCH + k0 + hi * 8;
    AF af;
#pragma unroll
    for (int i = 0; i < 8; ++i) {
      af.e[i]     = (_Float16)sp[i];
      af.e[8 + i] = (_Float16)sp[16 + i];
    }
#pragma unroll
    for (int ns = 0; ns < 4; ++ns) {
      const v16h bf = load_b_frag(vb + (size_t)(ns * 16 + l16) * SEQ + k0 + hi * 8);
      acc[ns] = __builtin_amdgcn_wmma_f32_16x16x32_f16(false, af.v, false, bf, (short)0, acc[ns], false, false);
    }
  }
#pragma unroll
  for (int ns = 0; ns < 4; ++ns)
#pragma unroll
    for (int r = 0; r < 8; ++r)
      atomicAdd(&O[(hi * 8 + r) * 64 + ns * 16 + l16], acc[ns][r]);
  __syncthreads();

  for (int i = threadIdx.x; i < 16 * 64; i += 256) {
    const int rr = i >> 6, cc = i & 63;
    o[((size_t)blockIdx.z * SEQ + m0 + rr) * EMBED + blockIdx.y * HD + cc] = (_Float16)O[i];
  }
}

// ---------------------------------------------------------------------------
extern "C" void kernel_launch(void* const* d_in, const int* in_sizes, int n_in,
                              void* d_out, int out_size, void* d_ws, size_t ws_size,
                              hipStream_t stream) {
  (void)in_sizes; (void)n_in; (void)out_size; (void)ws_size;
  const float* x     = (const float*)d_in[0];
  const float* ln1_g = (const float*)d_in[1];
  const float* ln1_b = (const float*)d_in[2];
  const float* qkv_w = (const float*)d_in[3];
  const float* out_w = (const float*)d_in[4];
  const float* out_b = (const float*)d_in[5];
  const float* ln2_g = (const float*)d_in[6];
  const float* ln2_b = (const float*)d_in[7];
  const float* fc1_w = (const float*)d_in[8];
  const float* fc1_b = (const float*)d_in[9];
  const float* fc2_w = (const float*)d_in[10];
  const float* fc2_b = (const float*)d_in[11];
  float* out = (float*)d_out;

  char* ws = (char*)d_ws;
  size_t off = 0;
  auto alloc = [&](size_t bytes) -> void* {
    void* p = ws + off;
    off = (off + bytes + 255) & ~(size_t)255;
    return p;
  };

  _Float16* h1   = (_Float16*)alloc((size_t)TOKENS * EMBED * 2);
  _Float16* wqkv = (_Float16*)alloc((size_t)QKV3 * EMBED * 2);
  _Float16* wout = (_Float16*)alloc((size_t)EMBED * EMBED * 2);
  _Float16* wfc1 = (_Float16*)alloc((size_t)HIDDEN * EMBED * 2);
  _Float16* wfc2 = (_Float16*)alloc((size_t)EMBED * HIDDEN * 2);
  _Float16* qkvb = (_Float16*)alloc((size_t)TOKENS * QKV3 * 2);
  _Float16* qf   = (_Float16*)alloc((size_t)TOKENS * EMBED * 2);
  _Float16* kf   = (_Float16*)alloc((size_t)TOKENS * EMBED * 2);
  _Float16* vTf  = (_Float16*)alloc((size_t)TOKENS * EMBED * 2);
  _Float16* of   = (_Float16*)alloc((size_t)TOKENS * EMBED * 2);
  float*    x1   = (float*)alloc((size_t)TOKENS * EMBED * 4);
  _Float16* h2   = (_Float16*)alloc((size_t)TOKENS * EMBED * 2);
  _Float16* h3   = (_Float16*)alloc((size_t)TOKENS * HIDDEN * 2);

  // weights -> f16 (recomputed every call: deterministic, no cached state)
  int n;
  n = QKV3 * EMBED;   cvt_f32_f16<<<(n + 255) / 256, 256, 0, stream>>>(qkv_w, wqkv, n);
  n = EMBED * EMBED;  cvt_f32_f16<<<(n + 255) / 256, 256, 0, stream>>>(out_w, wout, n);
  n = HIDDEN * EMBED; cvt_f32_f16<<<(n + 255) / 256, 256, 0, stream>>>(fc1_w, wfc1, n);
  n = EMBED * HIDDEN; cvt_f32_f16<<<(n + 255) / 256, 256, 0, stream>>>(fc2_w, wfc2, n);

  // h1 = LN1(x) in f16
  ln_to_f16<<<TOKENS, 256, 0, stream>>>(x, ln1_g, ln1_b, h1);

  // qkv = h1 @ qkv_w^T  (f16 out)
  gemm_f16_wmma<0><<<dim3(QKV3 / 64, TOKENS / 128), 128, 0, stream>>>(
      h1, wqkv, TOKENS, QKV3, EMBED, nullptr, nullptr, qkvb);

  {
    const int total = TOKENS * QKV3;
    split_qkv<<<(total + 255) / 256, 256, 0, stream>>>(qkvb, qf, kf, vTf);
  }

  // attention (S row tile kept in LDS: 16*2052 + 16*64 floats = 132.25 KB)
  const size_t attn_lds = (size_t)(16 * SPITCH + 16 * 64) * sizeof(float);
  (void)hipFuncSetAttribute((const void*)attn_kernel,
                            hipFuncAttributeMaxDynamicSharedMemorySize, (int)attn_lds);
  attn_kernel<<<dim3(SEQ / 16, HEADS, BATCH), 256, attn_lds, stream>>>(qf, kf, vTf, of);

  // x1 = x + o @ out_w^T + out_b   (f32)
  gemm_f16_wmma<1><<<dim3(EMBED / 64, TOKENS / 128), 128, 0, stream>>>(
      of, wout, TOKENS, EMBED, EMBED, out_b, x, x1);

  // h2 = LN2(x1) in f16
  ln_to_f16<<<TOKENS, 256, 0, stream>>>(x1, ln2_g, ln2_b, h2);

  // h3 = gelu(h2 @ fc1_w^T + fc1_b)  (f16)
  gemm_f16_wmma<2><<<dim3(HIDDEN / 64, TOKENS / 128), 128, 0, stream>>>(
      h2, wfc1, TOKENS, HIDDEN, EMBED, fc1_b, nullptr, h3);

  // out = x1 + h3 @ fc2_w^T + fc2_b  (f32)
  gemm_f16_wmma<1><<<dim3(EMBED / 64, TOKENS / 128), 128, 0, stream>>>(
      h3, wfc2, TOKENS, EMBED, HIDDEN, fc2_b, x1, out);
}
